// MoE_73194832658538
// MI455X (gfx1250) — matrix operand order repaired and checked by reading, not compiled, not run on hardware
//
#include <hip/hip_runtime.h>
#include <hip/hip_bf16.h>

// Problem sizes (fixed by the reference)
#define B_ 8192
#define D_ 1024
#define O_ 1024
#define E_ 8

typedef __bf16 bf16_t;
typedef __attribute__((ext_vector_type(16))) __bf16 v16bf;
typedef __attribute__((ext_vector_type(8)))  __bf16 v8bf;
typedef __attribute__((ext_vector_type(8)))  float   v8f;

// ---------------------------------------------------------------------------
// Kernel 1: split fp32 x into bf16 hi + bf16 lo (residual).  x ≈ hi + lo with
// ~16 effective mantissa bits when recombined through 3 WMMA products.
// ---------------------------------------------------------------------------
__global__ void moe_cvt_x(const float* __restrict__ x,
                          bf16_t* __restrict__ xhi,
                          bf16_t* __restrict__ xlo) {
  int i = blockIdx.x * blockDim.x + threadIdx.x;
  float v  = x[i];
  bf16_t h = (bf16_t)v;           // RNE f32->bf16
  xhi[i] = h;
  xlo[i] = (bf16_t)(v - (float)h);
}

// ---------------------------------------------------------------------------
// Kernel 2: transpose+convert We[e,d,o] (fp32) -> WeT[e,o,d] (bf16 hi/lo).
// Transposed storage makes the WMMA B-fragment a contiguous run in memory.
// LDS 32x33 tile keeps both read and write sides coalesced.
// ---------------------------------------------------------------------------
__global__ void moe_cvt_wet(const float* __restrict__ We,
                            bf16_t* __restrict__ wthi,
                            bf16_t* __restrict__ wtlo) {
  __shared__ float tile[32][33];
  const int e  = blockIdx.z;
  const int o0 = blockIdx.x * 32;
  const int d0 = blockIdx.y * 32;
  const int tx = threadIdx.x;           // 0..31
  const int ty = threadIdx.y;           // 0..7
  const float* src = We + (size_t)e * D_ * O_;
#pragma unroll
  for (int r = ty; r < 32; r += 8)
    tile[r][tx] = src[(size_t)(d0 + r) * O_ + (o0 + tx)];
  __syncthreads();
  bf16_t* dhi = wthi + (size_t)e * O_ * D_;
  bf16_t* dlo = wtlo + (size_t)e * O_ * D_;
#pragma unroll
  for (int r = ty; r < 32; r += 8) {
    float v = tile[tx][r];              // = We[e, d0+tx, o0+r]
    size_t oi = (size_t)(o0 + r) * D_ + (d0 + tx);
    bf16_t h = (bf16_t)v;
    dhi[oi] = h;
    dlo[oi] = (bf16_t)(v - (float)h);
  }
}

// ---------------------------------------------------------------------------
// Kernel 3: gating.  gates = softmax(x @ Wg + bg, axis=experts).
// Tiny (0.1% of the FLOPs): one wave32 per row, fp32 VALU, shuffle reduce.
// ---------------------------------------------------------------------------
__global__ void moe_gates(const float* __restrict__ x,
                          const float* __restrict__ Wg,
                          const float* __restrict__ bg,
                          float* __restrict__ gates) {
  const int wave = threadIdx.x >> 5;
  const int lane = threadIdx.x & 31;
  const int b    = blockIdx.x * 8 + wave;
  const float* xr = x + (size_t)b * D_;

  float acc[E_];
#pragma unroll
  for (int e = 0; e < E_; ++e) acc[e] = 0.f;

  for (int d = lane; d < D_; d += 32) {
    float xv = xr[d];
    const float4* wr = (const float4*)(Wg + (size_t)d * E_);
    float4 w0 = wr[0], w1 = wr[1];
    acc[0] += xv * w0.x;  acc[1] += xv * w0.y;
    acc[2] += xv * w0.z;  acc[3] += xv * w0.w;
    acc[4] += xv * w1.x;  acc[5] += xv * w1.y;
    acc[6] += xv * w1.z;  acc[7] += xv * w1.w;
  }
#pragma unroll
  for (int e = 0; e < E_; ++e)
#pragma unroll
    for (int off = 16; off >= 1; off >>= 1)
      acc[e] += __shfl_xor(acc[e], off, 32);

  float logit[E_], ex[E_];
  float m = -1e30f;
#pragma unroll
  for (int e = 0; e < E_; ++e) { logit[e] = acc[e] + bg[e]; m = fmaxf(m, logit[e]); }
  float s = 0.f;
#pragma unroll
  for (int e = 0; e < E_; ++e) { ex[e] = __expf(logit[e] - m); s += ex[e]; }
  float inv = 1.f / s;
#pragma unroll
  for (int e = 0; e < E_; ++e)
    if (lane == e) gates[(size_t)b * E_ + e] = ex[e] * inv;
}

// ---------------------------------------------------------------------------
// WMMA fragment loads (wave32, 16x16x32 bf16).
// A (16xK=32, ISA §7.12.2): lane m=l&15, half hs=l>>4;
//   elems 0..7  = A[m, k0 +  8*hs .. +7]
//   elems 8..15 = A[m, k0 + 16 + 8*hs .. +7]     -> two B128 loads, gap 16
// B (K=32 x 16), per sparse-B layout analogy: lane n=l&15, half hs;
//   elems 0..7  = B[k0 + 8*hs .. +7, n], elems 8..15 = B[k0 + 16 + 8*hs .. +7, n]
//                                                -> two B128 loads, gap 16
// ---------------------------------------------------------------------------
__device__ __forceinline__ v16bf load_frag2(const bf16_t* p, int gap) {
  v8bf a = *(const v8bf*)(p);
  v8bf b = *(const v8bf*)(p + gap);
  v16bf r;
#pragma unroll
  for (int i = 0; i < 8; ++i) { r[i] = a[i]; r[i + 8] = b[i]; }
  return r;
}

// ---------------------------------------------------------------------------
// Kernel 4: fused per-expert GEMM + gate-weighted combine.
// Block = 256 threads = 8 waves; wave w owns expert w for a 32(b) x 64(o)
// output tile (two 16x64 accumulator sets sharing every B fragment:
// 24 B128 loads : 24 v_wmma per k-step).  The 24 WMMAs are issued
// product-major (lo*hi pass, hi*lo pass, hi*hi pass over all 8 accumulators)
// so dependent WMMAs on the same accumulator are 8 issue slots apart --
// beyond the 5-slot bf16 WMMA->WMMA hazard window (ISA 7.12.1), eliminating
// the v_nop padding the adjacent-product ordering provoked.
// Epilogue folds be[e,o] and gates[b,e]; LDS sum over the 8 wave slices combines experts.
// ---------------------------------------------------------------------------
__launch_bounds__(256)
__global__ void moe_main(const bf16_t* __restrict__ xhi,
                         const bf16_t* __restrict__ xlo,
                         const bf16_t* __restrict__ wthi,
                         const bf16_t* __restrict__ wtlo,
                         const float* __restrict__ gates,
                         const float* __restrict__ be,
                         float* __restrict__ out) {
  __shared__ float red[8 * 32 * 64];       // 64 KB: one 32x64 slice per wave

  const int wave = threadIdx.x >> 5;       // == expert id
  const int lane = threadIdx.x & 31;
  const int hs   = lane >> 4;              // half-wave select
  const int l16  = lane & 15;
  const int e    = wave;
  const int o0   = blockIdx.x * 64;
  const int b0   = blockIdx.y * 32;

  // A-fragment row bases for the two m-tiles (hi and lo mantissa planes)
  const bf16_t* ah[2];
  const bf16_t* al[2];
#pragma unroll
  for (int m = 0; m < 2; ++m) {
    ah[m] = xhi + (size_t)(b0 + m * 16 + l16) * D_ + 8 * hs;
    al[m] = xlo + (size_t)(b0 + m * 16 + l16) * D_ + 8 * hs;
  }
  // B-fragment row bases for the four n-tiles of this wave's expert
  const bf16_t* bh[4];
  const bf16_t* bl[4];
#pragma unroll
  for (int n = 0; n < 4; ++n) {
    size_t row = (size_t)e * O_ + (o0 + n * 16 + l16);
    bh[n] = wthi + row * D_ + 8 * hs;
    bl[n] = wtlo + row * D_ + 8 * hs;
  }

  v8f acc[2][4];
#pragma unroll
  for (int m = 0; m < 2; ++m)
#pragma unroll
    for (int n = 0; n < 4; ++n)
      acc[m][n] = (v8f){0.f,0.f,0.f,0.f,0.f,0.f,0.f,0.f};

  for (int k0 = 0; k0 < D_; k0 += 32) {
    // Stage all fragments for this k-step first (12 x 2 B128 loads).
    v16bf fa_h[2], fa_l[2], fb_h[4], fb_l[4];
#pragma unroll
    for (int m = 0; m < 2; ++m) {
      fa_h[m] = load_frag2(ah[m] + k0, 16);
      fa_l[m] = load_frag2(al[m] + k0, 16);
    }
#pragma unroll
    for (int n = 0; n < 4; ++n) {
      fb_h[n] = load_frag2(bh[n] + k0, 16);
      fb_l[n] = load_frag2(bl[n] + k0, 16);
    }
    // Pass 1: lo(A) * hi(B) into all 8 accumulators.
#pragma unroll
    for (int m = 0; m < 2; ++m)
#pragma unroll
      for (int n = 0; n < 4; ++n)
        acc[m][n] = __builtin_amdgcn_wmma_f32_16x16x32_bf16(
            false, fa_l[m], false, fb_h[n], (short)0, acc[m][n], false, false);
    // Pass 2: hi(A) * lo(B).
#pragma unroll
    for (int m = 0; m < 2; ++m)
#pragma unroll
      for (int n = 0; n < 4; ++n)
        acc[m][n] = __builtin_amdgcn_wmma_f32_16x16x32_bf16(
            false, fa_h[m], false, fb_l[n], (short)0, acc[m][n], false, false);
    // Pass 3: hi(A) * hi(B).
#pragma unroll
    for (int m = 0; m < 2; ++m)
#pragma unroll
      for (int n = 0; n < 4; ++n)
        acc[m][n] = __builtin_amdgcn_wmma_f32_16x16x32_bf16(
            false, fa_h[m], false, fb_h[n], (short)0, acc[m][n], false, false);
  }

  // C/D layout: VGPR i, lanes<16 -> row i, lanes>=16 -> row i+8; col = l16.
  float g[2][8];
#pragma unroll
  for (int m = 0; m < 2; ++m)
#pragma unroll
    for (int i = 0; i < 8; ++i)
      g[m][i] = gates[(size_t)(b0 + m * 16 + i + 8 * hs) * E_ + e];

  float* slice = red + wave * (32 * 64);
#pragma unroll
  for (int n = 0; n < 4; ++n) {
    float bias = be[(size_t)e * O_ + (o0 + n * 16 + l16)];
#pragma unroll
    for (int m = 0; m < 2; ++m)
#pragma unroll
      for (int i = 0; i < 8; ++i) {
        float v = (acc[m][n][i] + bias) * g[m][i];
        slice[(m * 16 + i + 8 * hs) * 64 + n * 16 + l16] = v;
      }
  }
  __syncthreads();

  // Combine across experts: 2048 outputs, 256 threads x 8 contiguous each.
  const int t = threadIdx.x;
#pragma unroll
  for (int half = 0; half < 2; ++half) {
    const int idx = t * 8 + half * 4;
    float4 s = make_float4(0.f, 0.f, 0.f, 0.f);
#pragma unroll
    for (int w = 0; w < 8; ++w) {
      float4 v = *(const float4*)(red + w * (32 * 64) + idx);
      s.x += v.x; s.y += v.y; s.z += v.z; s.w += v.w;
    }
    const int row = idx >> 6;
    const int col = idx & 63;
    *(float4*)(out + (size_t)(b0 + row) * O_ + o0 + col) = s;
  }
}

// ---------------------------------------------------------------------------
// Launcher.  d_ws layout:
//   [0, 16MB)   xhi   (B*D bf16)
//   [16, 32MB)  xlo
//   [32, 48MB)  wthi  (E*O*D bf16, transposed)
//   [48, 64MB)  wtlo
//   [64MB, +256KB) gates (B*E f32)
// ---------------------------------------------------------------------------
extern "C" void kernel_launch(void* const* d_in, const int* in_sizes, int n_in,
                              void* d_out, int out_size, void* d_ws, size_t ws_size,
                              hipStream_t stream) {
  (void)in_sizes; (void)n_in; (void)out_size; (void)ws_size;
  const float* x  = (const float*)d_in[0];
  const float* Wg = (const float*)d_in[1];
  const float* bg = (const float*)d_in[2];
  const float* We = (const float*)d_in[3];
  const float* be = (const float*)d_in[4];
  float* out = (float*)d_out;

  bf16_t* xhi  = (bf16_t*)d_ws;
  bf16_t* xlo  = xhi  + (size_t)B_ * D_;
  bf16_t* wthi = xlo  + (size_t)B_ * D_;
  bf16_t* wtlo = wthi + (size_t)E_ * O_ * D_;
  float*  gates = (float*)(wtlo + (size_t)E_ * O_ * D_);

  moe_cvt_x  <<<(B_ * D_) / 256, 256, 0, stream>>>(x, xhi, xlo);
  moe_cvt_wet<<<dim3(O_ / 32, D_ / 32, E_), dim3(32, 8), 0, stream>>>(We, wthi, wtlo);
  moe_gates  <<<B_ / 8, 256, 0, stream>>>(x, Wg, bg, gates);
  moe_main   <<<dim3(O_ / 64, B_ / 32), 256, 0, stream>>>(xhi, xlo, wthi, wtlo,
                                                          gates, be, out);
}
